// PerJointHead_1297080123953
// MI455X (gfx1250) — matrix operand order repaired and checked by reading, not compiled, not run on hardware
//
#include <hip/hip_runtime.h>
#include <math.h>

// ---------------------------------------------------------------------------
// PerJointHead on MI455X (gfx1250):
//   LayerNorm(512) -> fc1(512x1024, per-joint) -> exact GELU -> fc2(1024x3)
// fc1 dominates (139 GFLOP, f16 WMMA w/ f32 accumulate). HBM floor ~12us
// (271MB fp32 x). fc1 weights (17.8MB f16-packed) are L2-resident; per-block
// B-fragment traffic scales as 1/rows_per_block -> 48 rows/block (2.75GB L2).
// Round-4 fix: round-3's compiler hoisted all 48 t-invariant A fragments out
// of the t loop and spilled them to scratch (scratch reloads in the hot
// loop). A fragments are now loaded with asm-volatile ds_load_b128 + explicit
// s_wait_dscnt inside the kc body, so exactly 3 fragments are live at a time.
// ---------------------------------------------------------------------------

typedef __attribute__((ext_vector_type(16))) _Float16     v16h;
typedef __attribute__((ext_vector_type(8)))  float        v8f;
typedef __attribute__((ext_vector_type(4)))  float        v4f;
typedef __attribute__((ext_vector_type(2)))  float        v2f;
typedef __attribute__((ext_vector_type(4)))  unsigned int v4u;

#define NJ 17
#define DD 512
#define HH 1024
#define ROWS_PER_BLK 48   // 3 M-tiles; 7776 = 162 * 48
#define MT 3
#define ASTR 520          // 512 + 8 halfs pad -> LDS row stride 1040B (65*16B)
#define KCHUNKS (DD / 32) // 16
#define NTILES  (HH / 16) // 64

union FragU { v4u u[2]; v16h h; };
union HalfPack { _Float16 h[2]; unsigned int u; };

// ---------------------------------------------------------------------------
// Pack fc1_w (fp32 [J][512][1024]) into f16 WMMA B-fragments:
//   layout [j][kc(16)][nt(64)][lane(32)][v(8)] dwords; a lane's 8 dwords are
//   contiguous -> main kernel loads each fragment as 2 coalesced b128.
//   For v_wmma_f32_16x16x32_f16: lane (0..15)+16*khi holds n=lane%16,
//   k = kc*32 + khi*16 + 2v .. +1 (packed half2 per dword).
// ---------------------------------------------------------------------------
__global__ __launch_bounds__(256)
void pack_w1_kernel(const float* __restrict__ w1, unsigned int* __restrict__ wb)
{
    unsigned int i = blockIdx.x * 256u + threadIdx.x;      // packed dword index
    unsigned int v    = i & 7u;
    unsigned int lane = (i >> 3) & 31u;
    unsigned int nt   = (i >> 8) & 63u;
    unsigned int kc   = (i >> 14) & 15u;
    unsigned int j    = i >> 18;
    if (j >= NJ) return;                                   // grid is exact; safety
    unsigned int n = nt * 16u + (lane & 15u);
    unsigned int k = kc * 32u + (lane >> 4) * 16u + 2u * v;
    float f0 = w1[((size_t)j * DD + k)     * HH + n];
    float f1 = w1[((size_t)j * DD + k + 1) * HH + n];
    HalfPack p;
    p.h[0] = (_Float16)f0;
    p.h[1] = (_Float16)f1;
    wb[i] = p.u;
}

// ---------------------------------------------------------------------------
// Fused head kernel. Block = 48 rows of one joint. 256 threads = 8 waves.
// Wave owns 8 N-tiles (8 waves x 8 x 16 = H), processed one at a time
// (outer t loop): per tile, full K loop over 3 M-tiles sharing each B
// fragment, then the tile's GELU+fc2 contribution folds into py registers.
// ---------------------------------------------------------------------------
__global__ __launch_bounds__(256)
void head_kernel(const float* __restrict__ x,
                 const float* __restrict__ gamma,
                 const float* __restrict__ beta,
                 const unsigned int* __restrict__ wb,   // packed f16 fc1_w
                 const float* __restrict__ b1,          // [J][H]
                 const float* __restrict__ w2,          // [J][H][3]
                 const float* __restrict__ b2,          // [J][3]
                 float* __restrict__ out)               // [M][J][3]
{
    __shared__ __align__(16) _Float16 Ah[ROWS_PER_BLK][ASTR];  // 50 KB
    __shared__ float mu_s[ROWS_PER_BLK];
    __shared__ float rs_s[ROWS_PER_BLK];
    __shared__ float yacc[ROWS_PER_BLK][3];

    const int tid  = threadIdx.x;
    const int wave = tid >> 5;
    const int lane = tid & 31;
    const int m16  = lane & 15;
    const int khi  = lane >> 4;
    const int j    = blockIdx.y;
    const int row0 = blockIdx.x * ROWS_PER_BLK;

    // seed y accumulator with fc2 bias (covered by the barrier below)
    if (tid < ROWS_PER_BLK * 3) {
        yacc[tid / 3][tid % 3] = b2[j * 3 + (tid % 3)];
    }

    // ---- LayerNorm statistics straight from global ------------------------
    // wave w handles rows 6w..6w+5; each lane owns a contiguous 16-float
    // segment (4x float4, coalesced 2KB per row per wave).
    for (int rr = 0; rr < 6; ++rr) {
        int r = wave * 6 + rr;
        const float* xr = x + (((size_t)(row0 + r)) * NJ + j) * DD;
        const v4f* xv = (const v4f*)xr + lane * 4;
        float s = 0.f, q = 0.f;
        #pragma unroll
        for (int i = 0; i < 4; ++i) {
            v4f u = xv[i];
            s += u.x + u.y + u.z + u.w;
            q += u.x * u.x + u.y * u.y + u.z * u.z + u.w * u.w;
        }
        #pragma unroll
        for (int off = 16; off > 0; off >>= 1) {
            s += __shfl_xor(s, off, 32);
            q += __shfl_xor(q, off, 32);
        }
        if (lane == 0) {
            float mu  = s * (1.f / DD);
            float var = q * (1.f / DD) - mu * mu;
            mu_s[r] = mu;
            rs_s[r] = rsqrtf(var + 1e-5f);
        }
    }
    __syncthreads();

    // ---- normalize + affine -> f16 A tile in LDS (x re-read, L2-hot) ------
    for (int i = tid; i < ROWS_PER_BLK * (DD / 2); i += 256) {
        int r  = i >> 8;          // DD/2 = 256 half2 per row
        int c2 = i & 255;
        int k  = c2 * 2;
        const float* xr = x + (((size_t)(row0 + r)) * NJ + j) * DD;
        v2f xv = *(const v2f*)&xr[k];
        v2f gv = *(const v2f*)&gamma[k];
        v2f bv = *(const v2f*)&beta[k];
        float mu = mu_s[r], rs = rs_s[r];
        HalfPack p;
        p.h[0] = (_Float16)((xv.x - mu) * rs * gv.x + bv.x);
        p.h[1] = (_Float16)((xv.y - mu) * rs * gv.y + bv.y);
        *(unsigned int*)&Ah[r][k] = p.u;
    }
    __syncthreads();

    // ---- fc1 WMMA + fused GELU/fc2, one N-tile at a time ------------------
    // base of this joint's packed B fragments: [kc][nt][lane][8] dwords
    const unsigned int* wbj = wb + ((size_t)j * KCHUNKS) * (NTILES * 32 * 8);

    // LDS byte offsets of this lane's A-fragment bases for the 3 M-tiles
    // (contiguous k order, same as the B pack: lane m=lane%16, khi=lane/16 holds halfs
    //  [kc*32 + khi*16 .. +15] -> two b128, 16B apart).
    unsigned aoff0 = (unsigned)(uintptr_t)&Ah[0 * 16 + m16][khi * 16];
    unsigned aoff1 = (unsigned)(uintptr_t)&Ah[1 * 16 + m16][khi * 16];
    unsigned aoff2 = (unsigned)(uintptr_t)&Ah[2 * 16 + m16][khi * 16];

    float py[MT][8][3];
    #pragma unroll
    for (int mt = 0; mt < MT; ++mt)
        #pragma unroll
        for (int v = 0; v < 8; ++v)
            py[mt][v][0] = py[mt][v][1] = py[mt][v][2] = 0.f;

    for (int t = 0; t < 8; ++t) {          // wave's n-tile: nt = wave*8 + t
        const int nt = wave * 8 + t;

        v8f acc[MT];
        const v8f z8 = {0.f, 0.f, 0.f, 0.f, 0.f, 0.f, 0.f, 0.f};
        #pragma unroll
        for (int mt = 0; mt < MT; ++mt) acc[mt] = z8;

        // prime B pipeline with kc=0 for this n-tile
        FragU bcur;
        {
            const v4u* bp = (const v4u*)(wbj + (((size_t)nt) * 32 + lane) * 8);
            bcur.u[0] = bp[0];
            bcur.u[1] = bp[1];
        }

        for (int kc = 0; kc < KCHUNKS; ++kc) {
            // issue next B-fragment loads before using the current one
            const int kc1 = (kc < KCHUNKS - 1) ? kc + 1 : kc;
            const v4u* nb = (const v4u*)(wbj + (size_t)kc1 * (NTILES * 32 * 8)
                                             + (((size_t)nt) * 32 + lane) * 8);
            FragU bnxt;
            bnxt.u[0] = nb[0];
            bnxt.u[1] = nb[1];

            // A fragments via asm ds_load_b128 (volatile: cannot be hoisted
            // across the t loop -> no scratch spills; exactly 3 frags live).
            const unsigned kb = (unsigned)(kc * 64);   // kc*32 halfs = 64B
            FragU a0, a1, a2;
            asm volatile("ds_load_b128 %0, %2\n\t"
                         "ds_load_b128 %1, %2 offset:16"
                         : "=v"(a0.u[0]), "=v"(a0.u[1])
                         : "v"(aoff0 + kb) : "memory");
            asm volatile("ds_load_b128 %0, %2\n\t"
                         "ds_load_b128 %1, %2 offset:16"
                         : "=v"(a1.u[0]), "=v"(a1.u[1])
                         : "v"(aoff1 + kb) : "memory");
            asm volatile("ds_load_b128 %0, %2\n\t"
                         "ds_load_b128 %1, %2 offset:16"
                         : "=v"(a2.u[0]), "=v"(a2.u[1])
                         : "v"(aoff2 + kb) : "memory");
            // wait for DScnt==0; "+v" ties route the fragments through this
            // asm so the WMMAs below cannot be scheduled before the wait.
            asm volatile("s_wait_dscnt 0x0"
                         : "+v"(a0.u[0]), "+v"(a0.u[1]),
                           "+v"(a1.u[0]), "+v"(a1.u[1]),
                           "+v"(a2.u[0]), "+v"(a2.u[1]));

            acc[0] = __builtin_amdgcn_wmma_f32_16x16x32_f16(
                false, a0.h, false, bcur.h, (short)0, acc[0], false, false);
            acc[1] = __builtin_amdgcn_wmma_f32_16x16x32_f16(
                false, a1.h, false, bcur.h, (short)0, acc[1], false, false);
            acc[2] = __builtin_amdgcn_wmma_f32_16x16x32_f16(
                false, a2.h, false, bcur.h, (short)0, acc[2], false, false);
            bcur = bnxt;
        }

        // fold this n-tile into the fc2 partials (reduction over n = in-reg)
        // C layout: acc[mt][v] is h[m][n], m = mt*16 + v + 8*khi,
        //           n = nt*16 + m16 (lane-resident -> w2 scalars per lane).
        const int n = nt * 16 + m16;
        const float bias1 = b1[j * HH + n];
        const float* w2n = &w2[((size_t)j * HH + n) * 3];
        const float w20 = w2n[0], w21 = w2n[1], w22 = w2n[2];
        #pragma unroll
        for (int mt = 0; mt < MT; ++mt) {
            #pragma unroll
            for (int v = 0; v < 8; ++v) {
                float hv = acc[mt][v] + bias1;
                float g  = 0.5f * hv * (1.f + erff(hv * 0.70710678118654752f));
                py[mt][v][0] += g * w20;
                py[mt][v][1] += g * w21;
                py[mt][v][2] += g * w22;
            }
        }
    }

    // ---- cross-wave fc2 reduction in LDS ----------------------------------
    #pragma unroll
    for (int mt = 0; mt < MT; ++mt)
        #pragma unroll
        for (int v = 0; v < 8; ++v) {
            int m = mt * 16 + v + 8 * khi;
            atomicAdd(&yacc[m][0], py[mt][v][0]);
            atomicAdd(&yacc[m][1], py[mt][v][1]);
            atomicAdd(&yacc[m][2], py[mt][v][2]);
        }
    __syncthreads();

    if (tid < ROWS_PER_BLK * 3) {
        int m = tid / 3, o = tid % 3;
        out[(((size_t)(row0 + m)) * NJ + j) * 3 + o] = yacc[m][o];
    }
}

// ---------------------------------------------------------------------------
extern "C" void kernel_launch(void* const* d_in, const int* in_sizes, int n_in,
                              void* d_out, int out_size, void* d_ws, size_t ws_size,
                              hipStream_t stream)
{
    (void)in_sizes; (void)n_in; (void)out_size; (void)ws_size;
    const float* x     = (const float*)d_in[0];
    const float* gamma = (const float*)d_in[1];
    const float* beta  = (const float*)d_in[2];
    const float* w1    = (const float*)d_in[3];
    const float* b1    = (const float*)d_in[4];
    const float* w2    = (const float*)d_in[5];
    const float* b2    = (const float*)d_in[6];
    float* out = (float*)d_out;

    unsigned int* wb = (unsigned int*)d_ws;  // 17*512*1024 halfs = ~17.8 MB

    // 1) pack fc1 weights into f16 WMMA B-fragment layout
    const int packDwords = NJ * KCHUNKS * NTILES * 32 * 8;   // 4,456,448
    pack_w1_kernel<<<packDwords / 256, 256, 0, stream>>>(w1, wb);

    // 2) fused LN -> fc1(WMMA f16) -> GELU -> fc2
    dim3 grid(7776 / ROWS_PER_BLK, NJ, 1);   // 162 x 17
    head_kernel<<<grid, 256, 0, stream>>>(x, gamma, beta, wb, b1, w2, b2, out);
}
